// ReSFU_77120432767180
// MI455X (gfx1250) — matrix operand order repaired and checked by reading, compiled, not run on hardware
//
#include <hip/hip_runtime.h>
#include <hip/hip_bf16.h>

// ---------------------------------------------------------------------------
// ReSFU forward for MI455X (gfx1250, wave32, WMMA).
//  - channel-mixing convs = v_wmma_f32_16x16x32_f16 GEMMs; GN folded into
//    B-operand loads as a single FMA (stats stored as scale/shift)
//  - weights pre-swizzled into WMMA A-fragment f16 layout (b128 loads, no cvt)
//  - GEMM kernels templated on compile-time dims so channel strides fold into
//    VMEM immediate offsets (no per-channel address registers, no spills)
//  - PCDC intermediates stored f16 (halves the dominant v-buffer traffic)
// ---------------------------------------------------------------------------

typedef __attribute__((ext_vector_type(16))) _Float16 v16h_t;
typedef __attribute__((ext_vector_type(8)))  float    v8f_t;

union V16H { v16h_t v; _Float16 h[16]; };
union V8F  { v8f_t  v; float    f[8];  };

#define TB 256
#define WAVES_PER_BLOCK 8

#define WMMA_F16(A, B, C) \
  __builtin_amdgcn_wmma_f32_16x16x32_f16(false, (A), false, (B), (short)0, (C), false, false)

// ------------- group-norm statistics -> (scale, shift) per group ------------
// scale = rsqrt(var+eps), shift = -mean*scale  (normalize = x*scale + shift)
__global__ void k_gn_stats(const float* __restrict__ src, float* __restrict__ stats,
                           int C, int cpg, int HW, float eps) {
  const int ngroups = C / cpg;
  const int bg = blockIdx.x;
  const int b = bg / ngroups, g = bg % ngroups;
  const float* base = src + ((size_t)b * C + (size_t)g * cpg) * (size_t)HW;
  const long n = (long)cpg * HW;
  float s = 0.f, ss = 0.f;
  for (long t = threadIdx.x; t < n; t += blockDim.x) {
    float v = base[t]; s += v; ss += v * v;
  }
  __shared__ float sh0[TB], sh1[TB];
  sh0[threadIdx.x] = s; sh1[threadIdx.x] = ss;
  __syncthreads();
  for (int off = TB / 2; off > 0; off >>= 1) {
    if ((int)threadIdx.x < off) {
      sh0[threadIdx.x] += sh0[threadIdx.x + off];
      sh1[threadIdx.x] += sh1[threadIdx.x + off];
    }
    __syncthreads();
  }
  if (threadIdx.x == 0) {
    float mean = sh0[0] / (float)n;
    float var  = sh1[0] / (float)n - mean * mean;
    float sc   = rsqrtf(var + eps);
    stats[bg * 2 + 0] = sc;
    stats[bg * 2 + 1] = -mean * sc;
  }
}

// f16-source variant (for the PCDC v-buffers)
__global__ void k_gn_stats_h(const _Float16* __restrict__ src, float* __restrict__ stats,
                             int C, int cpg, int HW, float eps) {
  const int ngroups = C / cpg;
  const int bg = blockIdx.x;
  const int b = bg / ngroups, g = bg % ngroups;
  const _Float16* base = src + ((size_t)b * C + (size_t)g * cpg) * (size_t)HW;
  const long n = (long)cpg * HW;
  float s = 0.f, ss = 0.f;
  for (long t = threadIdx.x; t < n; t += blockDim.x) {
    float v = (float)base[t]; s += v; ss += v * v;
  }
  __shared__ float sh0[TB], sh1[TB];
  sh0[threadIdx.x] = s; sh1[threadIdx.x] = ss;
  __syncthreads();
  for (int off = TB / 2; off > 0; off >>= 1) {
    if ((int)threadIdx.x < off) {
      sh0[threadIdx.x] += sh0[threadIdx.x + off];
      sh1[threadIdx.x] += sh1[threadIdx.x + off];
    }
    __syncthreads();
  }
  if (threadIdx.x == 0) {
    float mean = sh0[0] / (float)n;
    float var  = sh1[0] / (float)n - mean * mean;
    float sc   = rsqrtf(var + eps);
    stats[bg * 2 + 0] = sc;
    stats[bg * 2 + 1] = -mean * sc;
  }
}

// ------ pack f32 weight matrix into WMMA A-fragment f16 layout --------------
// dst[((mt*ksteps + ks)*32 + lane)*16 + e], contiguous-k A layout used here (B loads match):
//   row M = mt*16 + (lane&15); K = ks*32 + (lane>>4)*16 + e
__global__ void k_pack_afrag(const float* __restrict__ w, _Float16* __restrict__ dst,
                             int mtiles, int ksteps, int kvalid, int mvalid, int rowStride) {
  int idx = blockIdx.x * blockDim.x + threadIdx.x;
  int total = mtiles * ksteps * 32 * 16;
  if (idx >= total) return;
  int e = idx & 15;
  int lane = (idx >> 4) & 31;
  int f = idx >> 9;
  int ks = f % ksteps, mt = f / ksteps;
  int row = mt * 16 + (lane & 15);
  int kk = ks * 32 + (lane >> 4) * 16 + e;
  float v = (kk < kvalid && row < mvalid) ? w[row * rowStride + kk] : 0.f;
  dst[idx] = (_Float16)v;
}

// --------- 1x1 conv 256->32 WMMA GEMM, GN fused, 2 pixel-tiles/wave ---------
template <int HW>
__global__ void k_conv1x1_wmma(const float* __restrict__ src, const float* __restrict__ stats,
                               const _Float16* __restrict__ wfrag, const float* __restrict__ bias,
                               float* __restrict__ dst) {
  const int wave = threadIdx.x >> 5, lane = threadIdx.x & 31;
  const int laneHi = lane >> 4, laneM = lane & 15;
  const int groups32 = HW >> 5;                 // 32 pixels per wave
  const int t32 = blockIdx.x * WAVES_PER_BLOCK + wave;
  const int b = t32 / groups32;
  const int pix0 = (t32 % groups32) * 32 + laneM;
  const float* sb = src + (size_t)b * 256 * HW;
  const float* st = stats + b * 64;
  const v16h_t* af = (const v16h_t*)wfrag;
  V8F a0c[2], a1c[2];
  #pragma unroll
  for (int t = 0; t < 2; ++t)
    #pragma unroll
    for (int r = 0; r < 8; ++r) { a0c[t].f[r] = 0.f; a1c[t].f[r] = 0.f; }
  for (int ks = 0; ks < 8; ++ks) {
    V16H a0, a1;
    a0.v = af[ks * 32 + lane];                  // out rows 0..15
    a1.v = af[(8 + ks) * 32 + lane];            // out rows 16..31
    const int kbase = ks * 32 + laneHi * 16;    // first channel this lane loads
    const float* sbk = sb + (size_t)kbase * HW; // constant-HW: e*HW folds to imm
    if (ks < 7)
      __builtin_prefetch(sbk + (size_t)32 * HW + pix0, 0, 1);
    float sc0 = st[((kbase >> 3) + 0) * 2], sh0 = st[((kbase >> 3) + 0) * 2 + 1];
    float sc1 = st[((kbase >> 3) + 1) * 2], sh1 = st[((kbase >> 3) + 1) * 2 + 1];
    #pragma unroll
    for (int t = 0; t < 2; ++t) {
      const int pix = pix0 + t * 16;
      V16H bm;
      #pragma unroll
      for (int e = 0; e < 16; ++e) {            // fused GN: one fma + cvt
        float sc = (e < 8) ? sc0 : sc1, sh = (e < 8) ? sh0 : sh1;
        bm.h[e] = (_Float16)fmaf(sbk[(size_t)e * HW + pix], sc, sh);
      }
      a0c[t].v = WMMA_F16(a0.v, bm.v, a0c[t].v);
      a1c[t].v = WMMA_F16(a1.v, bm.v, a1c[t].v);
    }
  }
  #pragma unroll
  for (int t = 0; t < 2; ++t) {
    const int pix = pix0 + t * 16;
    #pragma unroll
    for (int r = 0; r < 8; ++r) {
      int m = r + laneHi * 8;
      dst[((size_t)b * 32 + m) * HW + pix]      = a0c[t].f[r] + bias[m];
      dst[((size_t)b * 32 + m + 16) * HW + pix] = a1c[t].f[r] + bias[m + 16];
    }
  }
}

// ---- PCDC fused conv: relu( dil-3x3(GN k) - kd*(GN q) + b ), K=72 pad 96 ---
template <int H, int W>
__global__ void k_pcdc_conv_wmma(const float* __restrict__ kf, const float* __restrict__ kst,
                                 const float* __restrict__ qf, const float* __restrict__ qst,
                                 const _Float16* __restrict__ wfrag, const float* __restrict__ bias,
                                 _Float16* __restrict__ v1) {
  constexpr int HW = H * W;
  const int wave = threadIdx.x >> 5, lane = threadIdx.x & 31;
  const int tilesPerGroup = HW >> 4;
  const int tile = blockIdx.x * WAVES_PER_BLOCK + wave;
  const int b = tile / (4 * tilesPerGroup);
  const int rem = tile % (4 * tilesPerGroup);
  const int g = rem / tilesPerGroup;
  const int laneHi = lane >> 4, laneM = lane & 15;
  const int pix = (rem % tilesPerGroup) * 16 + laneM;
  const int i = pix / W, j = pix % W;
  const float* kb = kf + (size_t)b * 32 * HW;
  const float* qb = qf + (size_t)b * 32 * HW;
  const float* ks = kst + b * 16;
  const float* qs = qst + b * 16;
  const v16h_t* af = (const v16h_t*)wfrag;
  V8F c0, c1;
  #pragma unroll
  for (int r = 0; r < 8; ++r) { c0.f[r] = 0.f; c1.f[r] = 0.f; }
  for (int kstep = 0; kstep < 3; ++kstep) {
    V16H a0, a1, bm;
    a0.v = af[((g * 2 + 0) * 3 + kstep) * 32 + lane];
    a1.v = af[((g * 2 + 1) * 3 + kstep) * 32 + lane];
    #pragma unroll
    for (int e = 0; e < 16; ++e) {
      int kk = kstep * 32 + laneHi * 16 + e;
      float val = 0.f;
      if (kk < 72) {
        int c = kk / 9, tap = kk - c * 9;
        int dp = tap / 3 - 1, dq = tap % 3 - 1;
        int ii = min(max(i + dp * 4, 0), H - 1);   // edge-pad, dilation 4
        int jj = min(max(j + dq * 4, 0), W - 1);
        int ch = g * 8 + c, sg = ch >> 2;
        float kv = fmaf(kb[(size_t)ch * HW + ii * W + jj], ks[sg * 2], ks[sg * 2 + 1]);
        float qv = fmaf(qb[(size_t)ch * HW + pix],         qs[sg * 2], qs[sg * 2 + 1]);
        val = kv - qv;                             // central-difference term
      }
      bm.h[e] = (_Float16)val;
    }
    c0.v = WMMA_F16(a0.v, bm.v, c0.v);
    c1.v = WMMA_F16(a1.v, bm.v, c1.v);
  }
  #pragma unroll
  for (int r = 0; r < 8; ++r) {
    int m = r + laneHi * 8;
    int oc0 = g * 32 + m, oc1 = oc0 + 16;
    v1[((size_t)b * 128 + oc0) * HW + pix] = (_Float16)fmaxf(c0.f[r] + bias[oc0], 0.f);
    v1[((size_t)b * 128 + oc1) * HW + pix] = (_Float16)fmaxf(c1.f[r] + bias[oc1], 0.f);
  }
}

// --------- grouped 1x1 (per group 32->32), K=32 exact, fused ReLU -----------
template <int HW>
__global__ void k_c1_wmma(const _Float16* __restrict__ v1, const _Float16* __restrict__ wfrag,
                          const float* __restrict__ bias, _Float16* __restrict__ v2) {
  const int wave = threadIdx.x >> 5, lane = threadIdx.x & 31;
  const int tilesPerGroup = HW >> 4;
  const int tile = blockIdx.x * WAVES_PER_BLOCK + wave;
  const int b = tile / (4 * tilesPerGroup);
  const int rem = tile % (4 * tilesPerGroup);
  const int g = rem / tilesPerGroup;
  const int laneHi = lane >> 4, laneM = lane & 15;
  const int pix = (rem % tilesPerGroup) * 16 + laneM;
  const _Float16* vb = v1 + (size_t)b * 128 * HW + (size_t)(g * 32 + laneHi * 16) * HW;
  const v16h_t* af = (const v16h_t*)wfrag;
  V8F c0, c1;
  #pragma unroll
  for (int r = 0; r < 8; ++r) { c0.f[r] = 0.f; c1.f[r] = 0.f; }
  V16H a0, a1, bm;
  a0.v = af[(g * 2 + 0) * 32 + lane];
  a1.v = af[(g * 2 + 1) * 32 + lane];
  #pragma unroll
  for (int e = 0; e < 16; ++e)
    bm.h[e] = vb[(size_t)e * HW + pix];
  c0.v = WMMA_F16(a0.v, bm.v, c0.v);
  c1.v = WMMA_F16(a1.v, bm.v, c1.v);
  #pragma unroll
  for (int r = 0; r < 8; ++r) {
    int m = r + laneHi * 8;
    int oc0 = g * 32 + m, oc1 = oc0 + 16;
    v2[((size_t)b * 128 + oc0) * HW + pix] = (_Float16)fmaxf(c0.f[r] + bias[oc0], 0.f);
    v2[((size_t)b * 128 + oc1) * HW + pix] = (_Float16)fmaxf(c1.f[r] + bias[oc1], 0.f);
  }
}

// --- 1x1 128->9 (M padded 16), K=128, GN(8 of 16) fused, 2 tiles/wave -------
template <int HW>
__global__ void k_c2_wmma(const _Float16* __restrict__ v2, const float* __restrict__ vst,
                          const _Float16* __restrict__ wfrag, const float* __restrict__ bias,
                          float* __restrict__ scores, int accum) {
  const int wave = threadIdx.x >> 5, lane = threadIdx.x & 31;
  const int laneHi = lane >> 4, laneM = lane & 15;
  const int groups32 = HW >> 5;                 // 32 pixels per wave
  const int t32 = blockIdx.x * WAVES_PER_BLOCK + wave;
  const int b = t32 / groups32;
  const int pix0 = (t32 % groups32) * 32 + laneM;
  const _Float16* vb = v2 + (size_t)b * 128 * HW;
  const float* st = vst + b * 16;
  const v16h_t* af = (const v16h_t*)wfrag;
  V8F acc[2];
  #pragma unroll
  for (int t = 0; t < 2; ++t)
    #pragma unroll
    for (int r = 0; r < 8; ++r) acc[t].f[r] = 0.f;
  for (int ksg = 0; ksg < 4; ++ksg) {
    V16H a0;
    a0.v = af[ksg * 32 + lane];
    const int kbase = ksg * 32 + laneHi * 16;
    const _Float16* vk = vb + (size_t)kbase * HW;
    float sc = st[(kbase >> 4) * 2], sh = st[(kbase >> 4) * 2 + 1];
    #pragma unroll
    for (int t = 0; t < 2; ++t) {
      const int pix = pix0 + t * 16;
      V16H bm;
      #pragma unroll
      for (int e = 0; e < 16; ++e)
        bm.h[e] = (_Float16)fmaf((float)vk[(size_t)e * HW + pix], sc, sh);
      acc[t].v = WMMA_F16(a0.v, bm.v, acc[t].v);
    }
  }
  #pragma unroll
  for (int t = 0; t < 2; ++t) {
    const int pix = pix0 + t * 16;
    #pragma unroll
    for (int r = 0; r < 8; ++r) {
      int m = r + laneHi * 8;
      if (m < 9) {
        float val = acc[t].f[r] + bias[m];
        size_t o = ((size_t)b * 9 + m) * HW + pix;
        scores[o] = accum ? (scores[o] + val) : val;
      }
    }
  }
}

// ---------------- generic bilinear resize (jax half-pixel) ------------------
__global__ void k_bilinear(const float* __restrict__ src, float* __restrict__ dst,
                           int BC, int hin, int win, int hout, int wout) {
  int idx = blockIdx.x * blockDim.x + threadIdx.x;
  int total = BC * hout * wout;
  if (idx >= total) return;
  int j = idx % wout, i = (idx / wout) % hout, p = idx / (wout * hout);
  float sy = (i + 0.5f) * (float)hin / (float)hout - 0.5f;
  float sx = (j + 0.5f) * (float)win / (float)wout - 0.5f;
  int y0 = (int)floorf(sy), x0 = (int)floorf(sx);
  float fy = sy - y0, fx = sx - x0;
  int y1 = min(max(y0 + 1, 0), hin - 1), x1 = min(max(x0 + 1, 0), win - 1);
  y0 = min(max(y0, 0), hin - 1); x0 = min(max(x0, 0), win - 1);
  const float* sp = src + (size_t)p * hin * win;
  float v00 = sp[y0 * win + x0], v01 = sp[y0 * win + x1];
  float v10 = sp[y1 * win + x0], v11 = sp[y1 * win + x1];
  dst[idx] = (1.f - fy) * ((1.f - fx) * v00 + fx * v01) + fy * ((1.f - fx) * v10 + fx * v11);
}

// ---------------- 3x3 Gaussian [1,2,1]/4 ⊗ itself, edge clamp ---------------
__global__ void k_gauss3x3(const float* __restrict__ src, float* __restrict__ dst,
                           int BC, int H, int W) {
  int idx = blockIdx.x * blockDim.x + threadIdx.x;
  int total = BC * H * W;
  if (idx >= total) return;
  int j = idx % W, i = (idx / W) % H, p = idx / (W * H);
  const float* sp = src + (size_t)p * H * W;
  float acc = 0.f;
  #pragma unroll
  for (int dy = -1; dy <= 1; ++dy) {
    float wy = (dy == 0) ? 0.5f : 0.25f;
    int ii = min(max(i + dy, 0), H - 1);
    #pragma unroll
    for (int dx = -1; dx <= 1; ++dx) {
      float wx = (dx == 0) ? 0.5f : 0.25f;
      int jj = min(max(j + dx, 0), W - 1);
      acc += wy * wx * sp[ii * W + jj];
    }
  }
  dst[idx] = acc;
}

// -------- GF: half-res downsample of (guide,inp), pack {g,i,g*i,g*g} --------
__global__ void k_gf_down_pack(const float* __restrict__ q, const float* __restrict__ kf,
                               float* __restrict__ packed, int BC, int Hl, int Wl) {
  int idx = blockIdx.x * blockDim.x + threadIdx.x;
  int total = BC * Hl * Wl;
  if (idx >= total) return;
  int j = idx % Wl, i = (idx / Wl) % Hl, p = idx / (Wl * Hl);
  int Wf = 2 * Wl;
  size_t fp = (size_t)p * (4 * Hl * Wl);     // full-res plane = 2Hl*2Wl
  const float* qp = q + fp;
  const float* kp = kf + fp;
  int i2 = 2 * i, j2 = 2 * j;
  float g  = 0.25f * (qp[i2 * Wf + j2] + qp[i2 * Wf + j2 + 1] +
                      qp[(i2 + 1) * Wf + j2] + qp[(i2 + 1) * Wf + j2 + 1]);
  float il = 0.25f * (kp[i2 * Wf + j2] + kp[i2 * Wf + j2 + 1] +
                      kp[(i2 + 1) * Wf + j2] + kp[(i2 + 1) * Wf + j2 + 1]);
  size_t plane = (size_t)Hl * Wl;
  size_t o = (size_t)p * 4 * plane + i * Wl + j;
  packed[o] = g; packed[o + plane] = il; packed[o + 2 * plane] = g * il; packed[o + 3 * plane] = g * g;
}

// ---------------- horizontal box sum, radius r, zero pad --------------------
__global__ void k_box_h(const float* __restrict__ src, float* __restrict__ dst,
                        int P, int Hl, int Wl, int r) {
  int idx = blockIdx.x * blockDim.x + threadIdx.x;
  int total = P * Hl * Wl;
  if (idx >= total) return;
  int j = idx % Wl, i = (idx / Wl) % Hl, p = idx / (Wl * Hl);
  const float* sp = src + ((size_t)p * Hl + i) * Wl;
  int lo = max(j - r, 0), hi = min(j + r, Wl - 1);
  float s = 0.f;
  for (int t = lo; t <= hi; ++t) s += sp[t];
  dst[idx] = s;
}

// --------- vertical box sum on 4 packed streams, then A,b per pixel ---------
__global__ void k_box_v_ab(const float* __restrict__ tmp, float* __restrict__ ab,
                           int BC, int Hl, int Wl, int r, float eps) {
  int idx = blockIdx.x * blockDim.x + threadIdx.x;
  int total = BC * Hl * Wl;
  if (idx >= total) return;
  int j = idx % Wl, i = (idx / Wl) % Hl, p = idx / (Wl * Hl);
  size_t plane = (size_t)Hl * Wl;
  const float* t0 = tmp + (size_t)p * 4 * plane;
  int lo = max(i - r, 0), hi = min(i + r, Hl - 1);
  float s0 = 0.f, s1 = 0.f, s2 = 0.f, s3 = 0.f;
  for (int t = lo; t <= hi; ++t) {
    size_t row = (size_t)t * Wl + j;
    s0 += t0[row]; s1 += t0[plane + row]; s2 += t0[2 * plane + row]; s3 += t0[3 * plane + row];
  }
  float N = (float)(hi - lo + 1) * (float)(min(j + r, Wl - 1) - max(j - r, 0) + 1);
  float mg = s0 / N, mi = s1 / N;
  float cov = s2 / N - mg * mi, var = s3 / N - mg * mg;
  float A = cov / (var + eps), bb = mi - A * mg;
  size_t o = (size_t)p * 2 * plane + i * Wl + j;
  ab[o] = A; ab[o + plane] = bb;
}

// ------------- upsample A,b bilinearly; q_gf = A*guide + b ------------------
__global__ void k_gf_up(const float* __restrict__ ab, const float* __restrict__ q,
                        float* __restrict__ qgf, int BC, int Hl, int Wl) {
  int H = 2 * Hl, W = 2 * Wl;
  int idx = blockIdx.x * blockDim.x + threadIdx.x;
  int total = BC * H * W;
  if (idx >= total) return;
  int j = idx % W, i = (idx / W) % H, p = idx / (W * H);
  float sy = (i + 0.5f) * 0.5f - 0.5f;
  float sx = (j + 0.5f) * 0.5f - 0.5f;
  int y0 = (int)floorf(sy), x0 = (int)floorf(sx);
  float fy = sy - y0, fx = sx - x0;
  int y1 = min(max(y0 + 1, 0), Hl - 1), x1 = min(max(x0 + 1, 0), Wl - 1);
  y0 = min(max(y0, 0), Hl - 1); x0 = min(max(x0, 0), Wl - 1);
  size_t plane = (size_t)Hl * Wl;
  const float* Ap = ab + (size_t)p * 2 * plane;
  const float* Bp = Ap + plane;
  float w00 = (1.f - fy) * (1.f - fx), w01 = (1.f - fy) * fx, w10 = fy * (1.f - fx), w11 = fy * fx;
  float A = w00 * Ap[y0 * Wl + x0] + w01 * Ap[y0 * Wl + x1] + w10 * Ap[y1 * Wl + x0] + w11 * Ap[y1 * Wl + x1];
  float bb = w00 * Bp[y0 * Wl + x0] + w01 * Bp[y0 * Wl + x1] + w10 * Bp[y1 * Wl + x0] + w11 * Bp[y1 * Wl + x1];
  qgf[idx] = A * q[idx] + bb;
}

// ------------- masked softmax over the 9 tap channels, in place -------------
__global__ void k_softmax_mask(float* __restrict__ sc, int Bn, int H, int W) {
  int idx = blockIdx.x * blockDim.x + threadIdx.x;
  int total = Bn * H * W;
  if (idx >= total) return;
  int j = idx % W, i = (idx / W) % H, b = idx / (W * H);
  size_t plane = (size_t)H * W;
  float s[9], mx = -1e30f;
  #pragma unroll
  for (int d = 0; d < 9; ++d) {
    int dp = d / 3, dq = d % 3;
    int u = i + 4 * dp - 4, v = j + 4 * dq - 4;
    float m = (u >= 0 && u < H && v >= 0 && v < W) ? 0.f : -1e7f;
    s[d] = sc[((size_t)b * 9 + d) * plane + i * W + j] + m;
    mx = fmaxf(mx, s[d]);
  }
  float sum = 0.f;
  #pragma unroll
  for (int d = 0; d < 9; ++d) { s[d] = __expf(s[d] - mx); sum += s[d]; }
  float inv = 1.f / sum;
  #pragma unroll
  for (int d = 0; d < 9; ++d) sc[((size_t)b * 9 + d) * plane + i * W + j] = s[d] * inv;
}

// ----------------- CARAFE reassembly against nearest-up x -------------------
__global__ void k_fns(const float* __restrict__ attn, const float* __restrict__ x,
                      float* __restrict__ out, int Bn, int H, int W, int hs, int wsd) {
  int idx = blockIdx.x * blockDim.x + threadIdx.x;
  int total = Bn * H * W;
  if (idx >= total) return;
  int j = idx % W, i = (idx / W) % H, b = idx / (W * H);
  size_t plane = (size_t)H * W;
  float aw[9]; int si[9];
  #pragma unroll
  for (int d = 0; d < 9; ++d) {
    int dp = d / 3, dq = d % 3;
    int u = i + 4 * dp - 4, v = j + 4 * dq - 4;
    bool ok = (u >= 0 && u < H && v >= 0 && v < W);
    aw[d] = ok ? attn[((size_t)b * 9 + d) * plane + i * W + j] : 0.f;
    si[d] = ok ? ((u >> 2) * wsd + (v >> 2)) : 0;
  }
  const float* xb = x + (size_t)b * 256 * hs * wsd;
  for (int c = 0; c < 256; ++c) {
    const float* xc = xb + (size_t)c * hs * wsd;
    float acc = 0.f;
    #pragma unroll
    for (int d = 0; d < 9; ++d) acc += aw[d] * xc[si[d]];
    out[((size_t)b * 256 + c) * plane + i * W + j] = acc;
  }
}

// ---------------------------------------------------------------------------
extern "C" void kernel_launch(void* const* d_in, const int* in_sizes, int n_in,
                              void* d_out, int out_size, void* d_ws, size_t ws_size,
                              hipStream_t stream) {
  (void)in_sizes; (void)n_in; (void)out_size; (void)ws_size;
  const float* y    = (const float*)d_in[0];
  const float* x    = (const float*)d_in[1];
  const float* qw   = (const float*)d_in[2];
  const float* qb   = (const float*)d_in[3];
  const float* kw   = (const float*)d_in[4];
  const float* kb   = (const float*)d_in[5];
  const float* sw   = (const float*)d_in[6];
  const float* sb   = (const float*)d_in[7];
  const float* sc1w = (const float*)d_in[8];
  const float* sc1b = (const float*)d_in[9];
  const float* sc2w = (const float*)d_in[10];
  const float* sc2b = (const float*)d_in[11];
  const float* dw   = (const float*)d_in[12];
  const float* db   = (const float*)d_in[13];
  const float* dc1w = (const float*)d_in[14];
  const float* dc1b = (const float*)d_in[15];
  const float* dc2w = (const float*)d_in[16];
  const float* dc2b = (const float*)d_in[17];
  float* out = (float*)d_out;

  constexpr int B = 2, H = 256, W = 256, HWl = H * W;
  constexpr int hs = 64, wsd = 64, HWs = hs * wsd;
  constexpr int Hl = 128, Wl = 128;

  // ---- byte-accurate workspace carving (all regions 256B aligned) ----
  char* p = (char*)d_ws;
  auto alloc = [&](size_t bytes) { char* r = p; p += (bytes + 255) & ~(size_t)255; return r; };
  const size_t SZ_Q = (size_t)B * 32 * HWl;                    // plane count
  float*    Q      = (float*)alloc(SZ_Q * 4);
  float*    KFULL  = (float*)alloc(SZ_Q * 4);
  float*    QGS    = (float*)alloc(SZ_Q * 4);
  float*    QGF    = (float*)alloc(SZ_Q * 4);
  float*    SCORES = (float*)alloc((size_t)B * 9 * HWl * 4);
  float*    KLOW   = (float*)alloc((size_t)B * 32 * HWs * 4);
  _Float16* V1H    = (_Float16*)alloc((size_t)B * 128 * HWl * 2);
  _Float16* V2H    = (_Float16*)alloc((size_t)B * 128 * HWl * 2);
  float*    STATS  = (float*)alloc(4096);
  _Float16* PQW    = (_Float16*)alloc(2 * 8 * 512 * 2);        // 1x1 q  (2 mt, 8 ks)
  _Float16* PKW    = (_Float16*)alloc(2 * 8 * 512 * 2);        // 1x1 k
  _Float16* PSW    = (_Float16*)alloc(8 * 3 * 512 * 2);        // pcdc s (8 mt, 3 ks)
  _Float16* PDW    = (_Float16*)alloc(8 * 3 * 512 * 2);        // pcdc d
  _Float16* PC1S   = (_Float16*)alloc(8 * 1 * 512 * 2);
  _Float16* PC1D   = (_Float16*)alloc(8 * 1 * 512 * 2);
  _Float16* PC2S   = (_Float16*)alloc(1 * 4 * 512 * 2);
  _Float16* PC2D   = (_Float16*)alloc(1 * 4 * 512 * 2);
  // guided-filter scratch aliases the (not yet live) f16 v-buffers
  float* GFP = (float*)V1H;                 // B*32*4*Hl*Wl f32 = 16.7MB
  float* GFT = GFP + SZ_Q;                  // 16.7MB  (V1H holds both exactly)
  float* ABL = (float*)V2H;                 // B*32*2*Hl*Wl f32 = 8.4MB

  float* ST_Y   = STATS;          // 2*32*2
  float* ST_X   = STATS + 128;
  float* ST_QGF = STATS + 256;    // 2*8*2
  float* ST_KF  = STATS + 320;
  float* ST_Q   = STATS + 384;
  float* ST_QGS = STATS + 448;
  float* ST_V2  = STATS + 512;

  // 0) pack all weight matrices into WMMA A-fragment f16 layout
  k_pack_afrag<<<(2 * 8 * 512 + TB - 1) / TB, TB, 0, stream>>>(qw,   PQW,  2, 8, 256,  32, 256);
  k_pack_afrag<<<(2 * 8 * 512 + TB - 1) / TB, TB, 0, stream>>>(kw,   PKW,  2, 8, 256,  32, 256);
  k_pack_afrag<<<(8 * 3 * 512 + TB - 1) / TB, TB, 0, stream>>>(sw,   PSW,  8, 3,  72, 128,  72);
  k_pack_afrag<<<(8 * 3 * 512 + TB - 1) / TB, TB, 0, stream>>>(dw,   PDW,  8, 3,  72, 128,  72);
  k_pack_afrag<<<(8 * 1 * 512 + TB - 1) / TB, TB, 0, stream>>>(sc1w, PC1S, 8, 1,  32, 128,  32);
  k_pack_afrag<<<(8 * 1 * 512 + TB - 1) / TB, TB, 0, stream>>>(dc1w, PC1D, 8, 1,  32, 128,  32);
  k_pack_afrag<<<(1 * 4 * 512 + TB - 1) / TB, TB, 0, stream>>>(sc2w, PC2S, 1, 4, 128,   9, 128);
  k_pack_afrag<<<(1 * 4 * 512 + TB - 1) / TB, TB, 0, stream>>>(dc2w, PC2D, 1, 4, 128,   9, 128);

  // 1) outer group-norm stats for y and x
  k_gn_stats<<<B * 32, TB, 0, stream>>>(y, ST_Y, 256, 8, HWl, 1e-5f);
  k_gn_stats<<<B * 32, TB, 0, stream>>>(x, ST_X, 256, 8, HWs, 1e-5f);

  // 2) q / k projections (WMMA GEMM, K=256, 2 pixel-tiles per wave)
  k_conv1x1_wmma<HWl><<<(B * HWl / 32) / WAVES_PER_BLOCK, TB, 0, stream>>>(y, ST_Y, PQW, qb, Q);
  k_conv1x1_wmma<HWs><<<(B * HWs / 32) / WAVES_PER_BLOCK, TB, 0, stream>>>(x, ST_X, PKW, kb, KLOW);

  // 3) k_ = bilinear 64->256 ; q_gs = gauss3x3(q)
  { int n = B * 32 * HWl; k_bilinear<<<(n + TB - 1) / TB, TB, 0, stream>>>(KLOW, KFULL, B * 32, hs, wsd, H, W); }
  { int n = B * 32 * HWl; k_gauss3x3<<<(n + TB - 1) / TB, TB, 0, stream>>>(Q, QGS, B * 32, H, W); }

  // 4) guided filter q_gf = GF(q, k_), half-res stats, r=8, eps=1e-3
  { int n = B * 32 * Hl * Wl;     k_gf_down_pack<<<(n + TB - 1) / TB, TB, 0, stream>>>(Q, KFULL, GFP, B * 32, Hl, Wl); }
  { int n = B * 32 * 4 * Hl * Wl; k_box_h<<<(n + TB - 1) / TB, TB, 0, stream>>>(GFP, GFT, B * 32 * 4, Hl, Wl, 8); }
  { int n = B * 32 * Hl * Wl;     k_box_v_ab<<<(n + TB - 1) / TB, TB, 0, stream>>>(GFT, ABL, B * 32, Hl, Wl, 8, 1e-3f); }
  { int n = B * 32 * HWl;         k_gf_up<<<(n + TB - 1) / TB, TB, 0, stream>>>(ABL, Q, QGF, B * 32, Hl, Wl); }

  // 5) PCDC pass 1: (k_, q_gf) -> SCORES (write)
  k_gn_stats<<<B * 8, TB, 0, stream>>>(QGF, ST_QGF, 32, 4, HWl, 1e-5f);
  k_gn_stats<<<B * 8, TB, 0, stream>>>(KFULL, ST_KF, 32, 4, HWl, 1e-5f);
  k_pcdc_conv_wmma<H, W><<<(B * 4 * HWl / 16) / WAVES_PER_BLOCK, TB, 0, stream>>>(KFULL, ST_KF, QGF, ST_QGF, PSW, sb, V1H);
  k_c1_wmma<HWl><<<(B * 4 * HWl / 16) / WAVES_PER_BLOCK, TB, 0, stream>>>(V1H, PC1S, sc1b, V2H);
  k_gn_stats_h<<<B * 8, TB, 0, stream>>>(V2H, ST_V2, 128, 16, HWl, 1e-5f);
  k_c2_wmma<HWl><<<(B * HWl / 32) / WAVES_PER_BLOCK, TB, 0, stream>>>(V2H, ST_V2, PC2S, sc2b, SCORES, 0);

  // 6) PCDC pass 2: (q_gs, q) -> SCORES (accumulate)
  k_gn_stats<<<B * 8, TB, 0, stream>>>(Q, ST_Q, 32, 4, HWl, 1e-5f);
  k_gn_stats<<<B * 8, TB, 0, stream>>>(QGS, ST_QGS, 32, 4, HWl, 1e-5f);
  k_pcdc_conv_wmma<H, W><<<(B * 4 * HWl / 16) / WAVES_PER_BLOCK, TB, 0, stream>>>(QGS, ST_QGS, Q, ST_Q, PDW, db, V1H);
  k_c1_wmma<HWl><<<(B * 4 * HWl / 16) / WAVES_PER_BLOCK, TB, 0, stream>>>(V1H, PC1D, dc1b, V2H);
  k_gn_stats_h<<<B * 8, TB, 0, stream>>>(V2H, ST_V2, 128, 16, HWl, 1e-5f);
  k_c2_wmma<HWl><<<(B * HWl / 32) / WAVES_PER_BLOCK, TB, 0, stream>>>(V2H, ST_V2, PC2D, dc2b, SCORES, 1);

  // 7) masked softmax over 9 taps, 8) CARAFE reassembly into d_out
  { int n = B * HWl; k_softmax_mask<<<(n + TB - 1) / TB, TB, 0, stream>>>(SCORES, B, H, W); }
  { int n = B * HWl; k_fns<<<(n + TB - 1) / TB, TB, 0, stream>>>(SCORES, x, out, B, H, W, hs, wsd); }
}
